// Attention_8048768712981
// MI455X (gfx1250) — matrix operand order repaired and checked by reading, not compiled, not run on hardware
//
#include <hip/hip_runtime.h>
#include <hip/hip_fp16.h>

typedef _Float16 h8  __attribute__((ext_vector_type(8)));
typedef _Float16 h16 __attribute__((ext_vector_type(16)));
typedef float    f8  __attribute__((ext_vector_type(8)));

// ---------------------------------------------------------------------------
// Fragment helpers (CDNA5 wave32 WMMA layouts, CDNA5 ISA §7.12.2)
// ---------------------------------------------------------------------------
static __device__ __forceinline__ h16 make_h16(h8 lo, h8 hi8) {
  h16 r;
#pragma unroll
  for (int i = 0; i < 8; ++i) { r[i] = lo[i]; r[i + 8] = hi8[i]; }
  return r;
}

// A-matrix 16x32 f16 fragment from a row-major f16 matrix (leading dim ld).
// lane l: row = base row + (l&15); k = hi*16 + e (contig)
static __device__ __forceinline__ h16 load_afrag(const _Float16* base, int ld,
                                                 int row, int k0, int hi) {
  const _Float16* p = base + (long)row * ld + k0 + hi * 16;
  return make_h16(*(const h8*)p, *(const h8*)(p + 8));
}

// B-matrix 32x16 f16 fragment from a TRANSPOSED (col-major-as-rows) matrix:
// bT[col][k], leading dim ldk. lane l: col = col index, k = k0 + hi*16 + e (contig)
static __device__ __forceinline__ h16 load_bfragT(const _Float16* bT, int ldk,
                                                  int col, int k0, int hi) {
  const _Float16* p = bT + (long)col * ldk + k0 + hi * 16;
  return make_h16(*(const h8*)p, *(const h8*)(p + 8));
}

static __device__ __forceinline__ f8 wmma16(h16 a, h16 b, f8 c) {
  return __builtin_amdgcn_wmma_f32_16x16x32_f16(false, a, false, b, (short)0, c,
                                                false, false);
}

// ---------------------------------------------------------------------------
// Kernel 1: window-partition x -> f16  AND  transpose+convert weights
// ---------------------------------------------------------------------------
__global__ __launch_bounds__(256) void k_prep(const float* __restrict__ x,
                                              const float* __restrict__ qkv_w,
                                              const float* __restrict__ proj_w,
                                              _Float16* __restrict__ xf,
                                              _Float16* __restrict__ wqkvT,
                                              _Float16* __restrict__ wprojT) {
  long i = (long)blockIdx.x * 256 + threadIdx.x;
  const long NX = 16384L * 384, NW1 = 1152L * 384, NW2 = 384L * 384;
  if (i < NX) {
    int c = (int)(i % 384);
    long t = i / 384;
    int tok = (int)(t & 511), w = (int)(t >> 9);
    int b = w >> 3, dblk = (w >> 2) & 1, hblk = (w >> 1) & 1, wblk = w & 1;
    int din = tok >> 6, hin = (tok >> 3) & 7, wq = tok & 7;
    int n = ((dblk * 8 + din) * 16 + hblk * 8 + hin) * 16 + wblk * 8 + wq;
    xf[i] = (_Float16)x[((long)b * 4096 + n) * 384 + c];
  } else if (i < NX + NW1) {
    long j = i - NX;
    int k = (int)(j % 384), n = (int)(j / 384);
    wqkvT[(long)n * 384 + k] = (_Float16)qkv_w[(long)k * 1152 + n];
  } else if (i < NX + NW1 + NW2) {
    long j = i - NX - NW1;
    int k = (int)(j % 384), n = (int)(j / 384);
    wprojT[(long)n * 384 + k] = (_Float16)proj_w[(long)k * 384 + n];
  }
}

// ---------------------------------------------------------------------------
// Kernel 2: QKV GEMM  [16384,384] x [384,1152] + bias, scatter to q/k/vT
//   q (raw, f16)  [win,head,tok,hd]
//   k (*scale)    [win,head,tok,hd]
//   vT            [win,head,hd,tok]
// One wave computes a 32x64 C tile, software-pipelined over K.
// ---------------------------------------------------------------------------
__global__ __launch_bounds__(256) void k_qkv(const _Float16* __restrict__ xf,
                                             const _Float16* __restrict__ wT,
                                             const float* __restrict__ bias,
                                             _Float16* __restrict__ qf,
                                             _Float16* __restrict__ kf,
                                             _Float16* __restrict__ vT,
                                             float scale) {
  int wid = ((int)blockIdx.x * 256 + (int)threadIdx.x) >> 5;
  int lane = threadIdx.x & 31, ln = lane & 15, hi = lane >> 4;
  int ntile = wid % 18, mtile = wid / 18;  // 18 n-tiles of 64, 512 m-tiles of 32
  int row0 = mtile * 32 + ln, row1 = row0 + 16;
  int nb = ntile * 64 + ln;
  f8 acc[2][4] = {};
  // prologue loads (k0 = 0)
  h16 a0 = load_afrag(xf, 384, row0, 0, hi);
  h16 a1 = load_afrag(xf, 384, row1, 0, hi);
  h16 b0 = load_bfragT(wT, 384, nb + 0, 0, hi);
  h16 b1 = load_bfragT(wT, 384, nb + 16, 0, hi);
  h16 b2 = load_bfragT(wT, 384, nb + 32, 0, hi);
  h16 b3 = load_bfragT(wT, 384, nb + 48, 0, hi);
#pragma unroll
  for (int k0 = 0; k0 < 384; k0 += 32) {
    h16 ca0 = a0, ca1 = a1, cb0 = b0, cb1 = b1, cb2 = b2, cb3 = b3;
    if (k0 + 32 < 384) {  // prefetch next step before this step's math
      a0 = load_afrag(xf, 384, row0, k0 + 32, hi);
      a1 = load_afrag(xf, 384, row1, k0 + 32, hi);
      b0 = load_bfragT(wT, 384, nb + 0, k0 + 32, hi);
      b1 = load_bfragT(wT, 384, nb + 16, k0 + 32, hi);
      b2 = load_bfragT(wT, 384, nb + 32, k0 + 32, hi);
      b3 = load_bfragT(wT, 384, nb + 48, k0 + 32, hi);
    }
    acc[0][0] = wmma16(ca0, cb0, acc[0][0]);
    acc[0][1] = wmma16(ca0, cb1, acc[0][1]);
    acc[0][2] = wmma16(ca0, cb2, acc[0][2]);
    acc[0][3] = wmma16(ca0, cb3, acc[0][3]);
    acc[1][0] = wmma16(ca1, cb0, acc[1][0]);
    acc[1][1] = wmma16(ca1, cb1, acc[1][1]);
    acc[1][2] = wmma16(ca1, cb2, acc[1][2]);
    acc[1][3] = wmma16(ca1, cb3, acc[1][3]);
  }
  int which = ntile / 6;  // 0=q 1=k 2=v (64 | 384)
#pragma unroll
  for (int rblk = 0; rblk < 2; ++rblk) {
    long t0 = (long)mtile * 32 + rblk * 16;
#pragma unroll
    for (int f = 0; f < 4; ++f) {
      int col = ntile * 64 + f * 16 + ln;
      int r = col % 384;
      int nh = r >> 6, hd = r & 63;
      float bv = bias[col];
#pragma unroll
      for (int j = 0; j < 8; ++j) {
        long t = t0 + j + hi * 8;
        int tok = (int)(t & 511), win = (int)(t >> 9);
        float v = acc[rblk][f][j] + bv;
        long hh = (long)win * 6 + nh;
        if (which == 0)
          qf[(hh * 512 + tok) * 64 + hd] = (_Float16)v;
        else if (which == 1)
          kf[(hh * 512 + tok) * 64 + hd] = (_Float16)(v * scale);
        else
          vT[(hh * 64 + hd) * 512 + tok] = (_Float16)v;
      }
    }
  }
}

// ---------------------------------------------------------------------------
// Kernel 3: decomposed relative-position bias tables
//   rel{x,y,z}[wh][qtok][k] = dot64(q[wh][qtok], rel_pos_{x,y,z}[qaxis-k+7])
// ---------------------------------------------------------------------------
__global__ __launch_bounds__(256) void k_rel(const _Float16* __restrict__ qf,
                                             const float* __restrict__ Rx,
                                             const float* __restrict__ Ry,
                                             const float* __restrict__ Rz,
                                             float* __restrict__ relx,
                                             float* __restrict__ rely,
                                             float* __restrict__ relz) {
  int i = (int)blockIdx.x * 256 + (int)threadIdx.x;  // wh*512 + qtok
  int qtok = i & 511, wh = i >> 9;
  int qd = qtok >> 6, qh = (qtok >> 3) & 7, qw = qtok & 7;
  const _Float16* qp = qf + ((long)wh * 512 + qtok) * 64;
  float ax[8] = {}, ay[8] = {}, az[8] = {};
  for (int c = 0; c < 64; ++c) {
    float qc = (float)qp[c];
#pragma unroll
    for (int k = 0; k < 8; ++k) {
      ax[k] += qc * Rx[(qd - k + 7) * 64 + c];
      ay[k] += qc * Ry[(qh - k + 7) * 64 + c];
      az[k] += qc * Rz[(qw - k + 7) * 64 + c];
    }
  }
  float* px = relx + (long)i * 8;
  float* py = rely + (long)i * 8;
  float* pz = relz + (long)i * 8;
#pragma unroll
  for (int k = 0; k < 8; ++k) { px[k] = ax[k]; py[k] = ay[k]; pz[k] = az[k]; }
}

// ---------------------------------------------------------------------------
// Kernel 4: flash attention per (window,head,16-query-block) per wave.
//   S = (q*scale)@k^T + bias ; online softmax ; O += P@V (via LDS re-layout)
//   V B-fragments are hoisted before the softmax to overlap VMEM with VALU.
// ---------------------------------------------------------------------------
__global__ __launch_bounds__(256) void k_attn(const _Float16* __restrict__ qf,
                                              const _Float16* __restrict__ kf,
                                              const _Float16* __restrict__ vT,
                                              const float* __restrict__ relx,
                                              const float* __restrict__ rely,
                                              const float* __restrict__ relz,
                                              _Float16* __restrict__ aout) {
  __shared__ __align__(16) _Float16 pbuf[8][16 * 32];  // per-wave P tile
  int wib = threadIdx.x >> 5;
  int wid = ((int)blockIdx.x * 256 + (int)threadIdx.x) >> 5;  // 0..6143
  int lane = threadIdx.x & 31, ln = lane & 15, hi = lane >> 4;
  int qblk = wid & 31, wh = wid >> 5;  // wh = win*6 + head in [0,192)

  const _Float16* qbase = qf + (long)wh * 512 * 64;
  const _Float16* kbase = kf + (long)wh * 512 * 64;
  const _Float16* vbase = vT + (long)wh * 64 * 512;

  int qrow = qblk * 16 + ln;
  h16 A0 = load_afrag(qbase, 64, qrow, 0, hi);
  h16 A1 = load_afrag(qbase, 64, qrow, 32, hi);

  // per-row state: VGPR j of a D tile is row (j + hi*8)
  float m[8], l[8];
  f8 O[4] = {};
#pragma unroll
  for (int j = 0; j < 8; ++j) { m[j] = -3.0e38f; l[j] = 0.0f; }

  long relbase = ((long)wh * 512 + qblk * 16 + hi * 8) * 8;
  const float* prx = relx + relbase;
  const float* pry = rely + relbase;
  const float* prz = relz + relbase;
  _Float16* pb = &pbuf[wib][0];

  for (int kb2 = 0; kb2 < 16; ++kb2) {
    int key0 = kb2 * 32;
    // ---- S tiles (two 16-key tiles) ----
    f8 s0 = {}, s1 = {};
    {
      int key = key0 + ln;
      const _Float16* kp = kbase + (long)key * 64 + hi * 16;
      s0 = wmma16(A0, make_h16(*(const h8*)kp, *(const h8*)(kp + 8)), s0);
      s0 = wmma16(A1, make_h16(*(const h8*)(kp + 32), *(const h8*)(kp + 40)), s0);
      const _Float16* kp1 = kp + 16 * 64;
      s1 = wmma16(A0, make_h16(*(const h8*)kp1, *(const h8*)(kp1 + 8)), s1);
      s1 = wmma16(A1, make_h16(*(const h8*)(kp1 + 32), *(const h8*)(kp1 + 40)), s1);
    }
    // ---- hoisted V B-fragments (independent of softmax below) ----
    h16 Bv[4];
#pragma unroll
    for (int f = 0; f < 4; ++f)
      Bv[f] = load_bfragT(vbase, 512, f * 16 + ln, key0, hi);
    // ---- relative position bias ----
    int kd0 = key0 >> 6, kd1 = (key0 + 16) >> 6;
    int kh0 = ((key0 + ln) >> 3) & 7, kh1 = ((key0 + 16 + ln) >> 3) & 7;
    int kw = ln & 7;
#pragma unroll
    for (int j = 0; j < 8; ++j) {
      float rzv = prz[j * 8 + kw];
      s0[j] += prx[j * 8 + kd0] + pry[j * 8 + kh0] + rzv;
      s1[j] += prx[j * 8 + kd1] + pry[j * 8 + kh1] + rzv;
    }
    // ---- online softmax over these 32 keys ----
#pragma unroll
    for (int j = 0; j < 8; ++j) {
      float t = fmaxf(s0[j], s1[j]);
#pragma unroll
      for (int d = 1; d < 16; d <<= 1) t = fmaxf(t, __shfl_xor(t, d, 32));
      float mn = fmaxf(m[j], t);
      float cf = __expf(m[j] - mn);
      float p0 = __expf(s0[j] - mn);
      float p1 = __expf(s1[j] - mn);
      float rs = p0 + p1;
#pragma unroll
      for (int d = 1; d < 16; d <<= 1) rs += __shfl_xor(rs, d, 32);
      l[j] = l[j] * cf + rs;
      m[j] = mn;
      s0[j] = p0;
      s1[j] = p1;
#pragma unroll
      for (int f = 0; f < 4; ++f) O[f][j] *= cf;
    }
    // ---- re-layout P (D layout) -> A fragment via wave-private LDS ----
#pragma unroll
    for (int j = 0; j < 8; ++j) {
      int r = j + hi * 8;
      pb[r * 32 + ln] = (_Float16)s0[j];
      pb[r * 32 + 16 + ln] = (_Float16)s1[j];
    }
    asm volatile("s_wait_dscnt 0" ::: "memory");
    h16 Ap = load_afrag(pb, 32, ln, 0, hi);
    // ---- O += P @ V ----
#pragma unroll
    for (int f = 0; f < 4; ++f) O[f] = wmma16(Ap, Bv[f], O[f]);
  }
  // ---- normalize and write [tok, head*64+hd] f16 for proj GEMM ----
  float inv[8];
#pragma unroll
  for (int j = 0; j < 8; ++j) inv[j] = 1.0f / l[j];
  int win = wh / 6, head = wh % 6;
#pragma unroll
  for (int f = 0; f < 4; ++f) {
    int hd = f * 16 + ln;
#pragma unroll
    for (int j = 0; j < 8; ++j) {
      int q = qblk * 16 + j + hi * 8;
      long t = (long)win * 512 + q;
      aout[t * 384 + head * 64 + hd] = (_Float16)(O[f][j] * inv[j]);
    }
  }
}

// ---------------------------------------------------------------------------
// Kernel 5: proj GEMM [16384,384]x[384,384] + bias, fused window-unpartition
// One wave computes a 32x64 C tile, software-pipelined over K.
// ---------------------------------------------------------------------------
__global__ __launch_bounds__(256) void k_proj(const _Float16* __restrict__ ain,
                                              const _Float16* __restrict__ wT,
                                              const float* __restrict__ bias,
                                              float* __restrict__ out) {
  int wid = ((int)blockIdx.x * 256 + (int)threadIdx.x) >> 5;
  int lane = threadIdx.x & 31, ln = lane & 15, hi = lane >> 4;
  int ntile = wid % 6, mtile = wid / 6;  // 6 n-tiles of 64, 512 m-tiles of 32
  int row0 = mtile * 32 + ln, row1 = row0 + 16;
  int nb = ntile * 64 + ln;
  f8 acc[2][4] = {};
  h16 a0 = load_afrag(ain, 384, row0, 0, hi);
  h16 a1 = load_afrag(ain, 384, row1, 0, hi);
  h16 b0 = load_bfragT(wT, 384, nb + 0, 0, hi);
  h16 b1 = load_bfragT(wT, 384, nb + 16, 0, hi);
  h16 b2 = load_bfragT(wT, 384, nb + 32, 0, hi);
  h16 b3 = load_bfragT(wT, 384, nb + 48, 0, hi);
#pragma unroll
  for (int k0 = 0; k0 < 384; k0 += 32) {
    h16 ca0 = a0, ca1 = a1, cb0 = b0, cb1 = b1, cb2 = b2, cb3 = b3;
    if (k0 + 32 < 384) {
      a0 = load_afrag(ain, 384, row0, k0 + 32, hi);
      a1 = load_afrag(ain, 384, row1, k0 + 32, hi);
      b0 = load_bfragT(wT, 384, nb + 0, k0 + 32, hi);
      b1 = load_bfragT(wT, 384, nb + 16, k0 + 32, hi);
      b2 = load_bfragT(wT, 384, nb + 32, k0 + 32, hi);
      b3 = load_bfragT(wT, 384, nb + 48, k0 + 32, hi);
    }
    acc[0][0] = wmma16(ca0, cb0, acc[0][0]);
    acc[0][1] = wmma16(ca0, cb1, acc[0][1]);
    acc[0][2] = wmma16(ca0, cb2, acc[0][2]);
    acc[0][3] = wmma16(ca0, cb3, acc[0][3]);
    acc[1][0] = wmma16(ca1, cb0, acc[1][0]);
    acc[1][1] = wmma16(ca1, cb1, acc[1][1]);
    acc[1][2] = wmma16(ca1, cb2, acc[1][2]);
    acc[1][3] = wmma16(ca1, cb3, acc[1][3]);
  }
#pragma unroll
  for (int rblk = 0; rblk < 2; ++rblk) {
#pragma unroll
    for (int f = 0; f < 4; ++f) {
      int col = ntile * 64 + f * 16 + ln;
      float bv = bias[col];
#pragma unroll
      for (int j = 0; j < 8; ++j) {
        long t = (long)mtile * 32 + rblk * 16 + j + hi * 8;
        int tok = (int)(t & 511), win = (int)(t >> 9);
        int b = win >> 3, dblk = (win >> 2) & 1, hblk = (win >> 1) & 1,
            wblk = win & 1;
        int din = tok >> 6, hin = (tok >> 3) & 7, wq = tok & 7;
        int n = ((dblk * 8 + din) * 16 + hblk * 8 + hin) * 16 + wblk * 8 + wq;
        out[((long)b * 4096 + n) * 384 + col] = acc[rblk][f][j] + bv;
      }
    }
  }
}

// ---------------------------------------------------------------------------
extern "C" void kernel_launch(void* const* d_in, const int* in_sizes, int n_in,
                              void* d_out, int out_size, void* d_ws,
                              size_t ws_size, hipStream_t stream) {
  const float* x = (const float*)d_in[0];
  const float* qkv_w = (const float*)d_in[1];
  const float* qkv_b = (const float*)d_in[2];
  const float* proj_w = (const float*)d_in[3];
  const float* proj_b = (const float*)d_in[4];
  const float* Rx = (const float*)d_in[5];
  const float* Ry = (const float*)d_in[6];
  const float* Rz = (const float*)d_in[7];

  char* ws = (char*)d_ws;
  size_t off = 0;
  auto carve = [&](size_t bytes) -> void* {
    void* p = ws + off;
    off = (off + bytes + 255) & ~(size_t)255;
    return p;
  };
  _Float16* xf = (_Float16*)carve(16384L * 384 * 2);
  _Float16* wqkvT = (_Float16*)carve(1152L * 384 * 2);
  _Float16* wprojT = (_Float16*)carve(384L * 384 * 2);
  _Float16* qf = (_Float16*)carve(192L * 512 * 64 * 2);
  _Float16* kf = (_Float16*)carve(192L * 512 * 64 * 2);
  _Float16* vT = (_Float16*)carve(192L * 64 * 512 * 2);
  float* relx = (float*)carve(192L * 512 * 8 * 4);
  float* rely = (float*)carve(192L * 512 * 8 * 4);
  float* relz = (float*)carve(192L * 512 * 8 * 4);
  _Float16* aout = (_Float16*)carve(16384L * 384 * 2);

  k_prep<<<26880, 256, 0, stream>>>(x, qkv_w, proj_w, xf, wqkvT, wprojT);
  k_qkv<<<1152, 256, 0, stream>>>(xf, wqkvT, qkv_b, qf, kf, vT, 0.125f);
  k_rel<<<384, 256, 0, stream>>>(qf, Rx, Ry, Rz, relx, rely, relz);
  k_attn<<<768, 256, 0, stream>>>(qf, kf, vT, relx, rely, relz, aout);
  k_proj<<<384, 256, 0, stream>>>(aout, wprojT, proj_b, (float*)d_out);
}
